// CausalAttention_45380624449901
// MI455X (gfx1250) — matrix operand order repaired and checked by reading, not compiled, not run on hardware
//
#include <hip/hip_runtime.h>

// Problem constants (match reference)
#define BB   4
#define SS   4096
#define DD   768

typedef __attribute__((ext_vector_type(16))) __bf16 v16bf;
typedef __attribute__((ext_vector_type(8)))  __bf16 v8bf;
typedef __attribute__((ext_vector_type(8)))  float  v8f;

// ---------------------------------------------------------------------------
// WMMA wrapper: D = A(16x32 bf16) * B(32x16 bf16) + C(16x16 f32)
// ---------------------------------------------------------------------------
__device__ __forceinline__ v8f wmma_bf16(v16bf a, v16bf b, v8f c) {
    return __builtin_amdgcn_wmma_f32_16x16x32_bf16(
        /*neg_a=*/false, a, /*neg_b=*/false, b,
        /*c_mod=*/(short)0, c, /*reuse_a=*/false, /*reuse_b=*/false);
}

// A-matrix fragment (16x32, 16-bit): lane L -> row M = L%16, half h = L/16.
// elements 0..7  -> K = h*8 + i ;  elements 8..15 -> K = 16 + h*8 + (i-8)
__device__ __forceinline__ v16bf load_a_bf16(const __bf16* p, int ld) {
    const int lane = threadIdx.x & 31;
    const int m = lane & 15, h = lane >> 4;
    const __bf16* r = p + m * ld + h * 8;
    v16bf a;
#pragma unroll
    for (int i = 0; i < 8; ++i) { a[i] = r[i]; a[8 + i] = r[16 + i]; }
    return a;
}

// B-matrix fragment (32x16, 16-bit): lane L -> column N = L%16, half h = L/16,
// elements 0..7  -> K = h*8 + i ;  elements 8..15 -> K = 16 + h*8 + (i-8)
// p points at logical [K=0][N=0] of an array stored row-major by N (row stride ld).
__device__ __forceinline__ v16bf load_b_bf16(const __bf16* p, int ld) {
    const int lane = threadIdx.x & 31;
    const int n = lane & 15, h = lane >> 4;
    const __bf16* r = p + (size_t)n * ld + h * 8;
    v16bf b;
#pragma unroll
    for (int i = 0; i < 8; ++i) { b[i] = r[i]; b[8 + i] = r[16 + i]; }
    return b;
}

// ---------------------------------------------------------------------------
// Kernel 1a: convert W (fp32 [DIN][DOUT]) -> Wt (bf16 [3][DOUT][DIN], transposed)
// ---------------------------------------------------------------------------
__global__ void k_convert_w(const float* __restrict__ Wq,
                            const float* __restrict__ Wk,
                            const float* __restrict__ Wv,
                            __bf16* __restrict__ Wt) {
    int idx = blockIdx.x * blockDim.x + threadIdx.x;
    const int total = 3 * DD * DD;
    if (idx >= total) return;
    const int mat  = idx / (DD * DD);
    const int rem  = idx % (DD * DD);
    const int dout = rem / DD;
    const int din  = rem % DD;
    const float* W = (mat == 0) ? Wq : (mat == 1) ? Wk : Wv;
    Wt[idx] = (__bf16)W[(size_t)din * DD + dout];
}

// ---------------------------------------------------------------------------
// Kernel 1b: convert x (fp32 [B*S][D]) -> Xb (bf16, same layout).
// 8 elements per thread: 2x float4 load -> 1x 16B bf16 store.
// ---------------------------------------------------------------------------
__global__ void k_convert_x(const float* __restrict__ x,
                            __bf16* __restrict__ Xb) {
    const size_t i = ((size_t)blockIdx.x * blockDim.x + threadIdx.x) * 8;
    const float4 a = *(const float4*)(x + i);
    const float4 b = *(const float4*)(x + i + 4);
    v8bf o;
    o[0] = (__bf16)a.x; o[1] = (__bf16)a.y; o[2] = (__bf16)a.z; o[3] = (__bf16)a.w;
    o[4] = (__bf16)b.x; o[5] = (__bf16)b.y; o[6] = (__bf16)b.z; o[7] = (__bf16)b.w;
    *(v8bf*)(Xb + i) = o;
}

// ---------------------------------------------------------------------------
// Kernel 2: QKV projections.  One wave = one 16(M) x 64(N) output strip
// (4 accumulators); all 4 B fragments are loaded before the 4 WMMAs so the
// scheduler can overlap matrix ops with in-flight loads (partial loadcnt waits).
//   Q,K -> bf16 row-major [B*S][D];  V -> bf16 transposed Vt[b][D][S]
// ---------------------------------------------------------------------------
__global__ void __launch_bounds__(128)
k_qkv_gemm(const __bf16* __restrict__ Xb,
           const __bf16* __restrict__ Wt,
           __bf16* __restrict__ Qb,
           __bf16* __restrict__ Kb,
           __bf16* __restrict__ Vt) {
    const int MT  = (BB * SS) / 16;  // 1024 row tiles
    const int NTB = DD / 64;         // 12 col strips of 64
    const int gid  = blockIdx.x * (blockDim.x >> 5) + (threadIdx.x >> 5);
    const int lane = threadIdx.x & 31;
    const int mat  = gid / (MT * NTB);
    const int rem  = gid % (MT * NTB);
    const int m0   = (rem / NTB) * 16;
    const int nb0  = (rem % NTB) * 64;

    const __bf16* W = Wt + (size_t)mat * DD * DD;   // [DOUT][DIN] transposed
    v8f c[4];
    v8f vzero = {};
#pragma unroll
    for (int j = 0; j < 4; ++j) c[j] = vzero;

    for (int d = 0; d < DD; d += 32) {
        v16bf a  = load_a_bf16(Xb + (size_t)m0 * DD + d, DD);
        v16bf b0 = load_b_bf16(W + (size_t)(nb0 +  0) * DD + d, DD);
        v16bf b1 = load_b_bf16(W + (size_t)(nb0 + 16) * DD + d, DD);
        v16bf b2 = load_b_bf16(W + (size_t)(nb0 + 32) * DD + d, DD);
        v16bf b3 = load_b_bf16(W + (size_t)(nb0 + 48) * DD + d, DD);
        c[0] = wmma_bf16(a, b0, c[0]);
        c[1] = wmma_bf16(a, b1, c[1]);
        c[2] = wmma_bf16(a, b2, c[2]);
        c[3] = wmma_bf16(a, b3, c[3]);
    }

    const int h = lane >> 4, n = lane & 15;
    if (mat == 2) {
        const int batch = m0 / SS;
        const int s0    = m0 % SS;   // 16-row tiles never cross batch boundary
#pragma unroll
        for (int j = 0; j < 4; ++j) {
            __bf16* vrow = Vt + ((size_t)batch * DD + nb0 + j * 16 + n) * SS;
#pragma unroll
            for (int r = 0; r < 8; ++r)
                vrow[s0 + h * 8 + r] = (__bf16)c[j][r];
        }
    } else {
        __bf16* O = (mat == 0) ? Qb : Kb;
#pragma unroll
        for (int j = 0; j < 4; ++j)
#pragma unroll
            for (int r = 0; r < 8; ++r)
                O[(size_t)(m0 + h * 8 + r) * DD + nb0 + j * 16 + n] = (__bf16)c[j][r];
    }
}

// ---------------------------------------------------------------------------
// Kernel 3: flash attention.  1 block (4 waves) per 16-query tile.
//   Block loop over 64-key blocks up to causal limit; online softmax in LDS.
//   Each wave: one 16x16 score tile, then 12 column tiles (192 cols) of O.
// ---------------------------------------------------------------------------
#define QS_LD (DD + 8)   // padded LDS stride to spread banks
#define SS_LD 68

__global__ void __launch_bounds__(128)
k_attn(const __bf16* __restrict__ Qb,
       const __bf16* __restrict__ Kb,
       const __bf16* __restrict__ Vt,
       float* __restrict__ out) {
    __shared__ __attribute__((aligned(16))) __bf16 Qs[16 * QS_LD]; // 24.3 KB
    __shared__ float  Ssc[16 * SS_LD];                             // scores
    __shared__ __attribute__((aligned(16))) __bf16 Pb[16 * 64];    // probs bf16
    __shared__ float  m_run[16], l_run[16], corr_s[16];

    const int batch = blockIdx.x >> 8;        // 256 q-tiles per batch
    const int q0    = (blockIdx.x & 255) * 16;
    const int tid   = threadIdx.x;
    const int wave  = tid >> 5, lane = tid & 31;
    const int n = lane & 15, h = lane >> 4;

    // Stage the 16xD Q tile into LDS (uint4 = 8 bf16 per copy).
    const __bf16* Qrow = Qb + ((size_t)batch * SS + q0) * DD;
    for (int i = tid; i < 16 * (DD / 8); i += 128) {
        const int r    = i / (DD / 8);
        const int cblk = (i % (DD / 8)) * 8;
        *(uint4*)(&Qs[r * QS_LD + cblk]) =
            *(const uint4*)(Qrow + (size_t)r * DD + cblk);
    }
    if (tid < 16) { m_run[tid] = -__builtin_inff(); l_run[tid] = 0.0f; }

    v8f o[12];
    v8f vzero = {};
#pragma unroll
    for (int t = 0; t < 12; ++t) o[t] = vzero;
    __syncthreads();

    const float scale = 0.03608439182435161f;  // 1/sqrt(768)
    const int qlast = q0 + 15;
    const int nblk  = (qlast >> 6) + 1;        // 64-key blocks to process

    const __bf16* Kbase = Kb + (size_t)batch * SS * DD;
    const __bf16* Vbase = Vt + (size_t)batch * DD * SS;

    for (int kb = 0; kb < nblk; ++kb) {
        const int kstart = kb * 64;
        const int ks     = kstart + wave * 16;   // this wave's 16 keys

        if (ks <= qlast) {                        // wave-uniform (EXEC full)
            if (ks + 64 < SS)                     // hint the next K block
                __builtin_prefetch(Kbase + (size_t)(ks + 64) * DD + lane * 64, 0, 0);
            const __bf16* Kp = Kbase + (size_t)ks * DD;
            v8f c = {};
            for (int d = 0; d < DD; d += 64) {    // 2 K-steps per iteration,
                v16bf a0 = load_a_bf16(Qs + d,      QS_LD);   // loads hoisted
                v16bf a1 = load_a_bf16(Qs + d + 32, QS_LD);
                v16bf b0 = load_b_bf16(Kp + d,      DD);
                v16bf b1 = load_b_bf16(Kp + d + 32, DD);
                c = wmma_bf16(a0, b0, c);
                c = wmma_bf16(a1, b1, c);
            }
#pragma unroll
            for (int r = 0; r < 8; ++r) {
                const int key = ks + n;
                const int qr  = q0 + h * 8 + r;
                const float s = (key <= qr) ? c[r] * scale : -__builtin_inff();
                Ssc[(h * 8 + r) * SS_LD + wave * 16 + n] = s;
            }
        } else {
#pragma unroll
            for (int r = 0; r < 8; ++r)
                Ssc[(h * 8 + r) * SS_LD + wave * 16 + n] = -__builtin_inff();
        }
        __syncthreads();

        // Online softmax row statistics (16 rows, one thread each).
        if (tid < 16) {
            const int row = tid;
            float mb = -__builtin_inff();
            for (int j = 0; j < 64; ++j) mb = fmaxf(mb, Ssc[row * SS_LD + j]);
            const float m_old = m_run[row];
            const float m_new = fmaxf(m_old, mb);      // finite after block 0
            const float cr    = __expf(m_old - m_new); // 0 on first block
            float sum = 0.0f;
            for (int j = 0; j < 64; ++j) {
                const float p = __expf(Ssc[row * SS_LD + j] - m_new);
                Pb[row * 64 + j] = (__bf16)p;
                sum += p;
            }
            m_run[row]  = m_new;
            l_run[row]  = l_run[row] * cr + sum;
            corr_s[row] = cr;
        }
        __syncthreads();

        // Rescale accumulators by the softmax correction, then O += P * V.
        float cv[8];
#pragma unroll
        for (int r = 0; r < 8; ++r) cv[r] = corr_s[h * 8 + r];
#pragma unroll
        for (int t = 0; t < 12; ++t)
#pragma unroll
            for (int r = 0; r < 8; ++r) o[t][r] *= cv[r];

#pragma unroll
        for (int f = 0; f < 2; ++f) {
            v16bf a = load_a_bf16(Pb + f * 32, 64);            // P fragment
            const __bf16* vb = Vbase + kstart + f * 32;
            if (kstart + 64 < SS)
                __builtin_prefetch(vb + (size_t)(wave * 192 + lane) * SS + 64, 0, 0);
#pragma unroll
            for (int tq = 0; tq < 3; ++tq) {       // groups of 4: hoist loads
                const int ncol = wave * 192 + tq * 64;
                v16bf b0 = load_b_bf16(vb + (size_t)(ncol +  0) * SS, SS);
                v16bf b1 = load_b_bf16(vb + (size_t)(ncol + 16) * SS, SS);
                v16bf b2 = load_b_bf16(vb + (size_t)(ncol + 32) * SS, SS);
                v16bf b3 = load_b_bf16(vb + (size_t)(ncol + 48) * SS, SS);
                o[tq * 4 + 0] = wmma_bf16(a, b0, o[tq * 4 + 0]);
                o[tq * 4 + 1] = wmma_bf16(a, b1, o[tq * 4 + 1]);
                o[tq * 4 + 2] = wmma_bf16(a, b2, o[tq * 4 + 2]);
                o[tq * 4 + 3] = wmma_bf16(a, b3, o[tq * 4 + 3]);
            }
        }
        __syncthreads();   // protect Ssc/Pb before next block
    }

    // Epilogue: divide by the softmax denominator and store fp32.
    float li[8];
#pragma unroll
    for (int r = 0; r < 8; ++r) li[r] = 1.0f / l_run[h * 8 + r];
    float* orow = out + ((size_t)batch * SS + q0) * DD;
#pragma unroll
    for (int t = 0; t < 12; ++t) {
        const int ncol = wave * 192 + t * 16;
#pragma unroll
        for (int r = 0; r < 8; ++r)
            orow[(size_t)(h * 8 + r) * DD + ncol + n] = o[t][r] * li[r];
    }
}

// ---------------------------------------------------------------------------
// Host launcher
// ---------------------------------------------------------------------------
extern "C" void kernel_launch(void* const* d_in, const int* in_sizes, int n_in,
                              void* d_out, int out_size, void* d_ws, size_t ws_size,
                              hipStream_t stream) {
    const float* x  = (const float*)d_in[0];
    const float* Wq = (const float*)d_in[1];
    const float* Wk = (const float*)d_in[2];
    const float* Wv = (const float*)d_in[3];
    float* out = (float*)d_out;

    // Workspace layout (all offsets 256B aligned)
    char* ws = (char*)d_ws;
    constexpr size_t WT_BYTES = (size_t)3 * DD * DD * 2;       //  3,538,944
    constexpr size_t XB_BYTES = (size_t)BB * SS * DD * 2;      // 25,165,824
    constexpr size_t QB_BYTES = XB_BYTES;
    __bf16* Wt = (__bf16*)(ws);
    __bf16* Xb = (__bf16*)(ws + WT_BYTES);
    __bf16* Qb = (__bf16*)(ws + WT_BYTES + XB_BYTES);
    __bf16* Kb = (__bf16*)(ws + WT_BYTES + XB_BYTES + QB_BYTES);
    __bf16* Vt = (__bf16*)(ws + WT_BYTES + XB_BYTES + 2 * QB_BYTES);

    // 1a) Weight convert + transpose
    {
        const int total = 3 * DD * DD;
        k_convert_w<<<(total + 255) / 256, 256, 0, stream>>>(Wq, Wk, Wv, Wt);
    }
    // 1b) Activation convert: 12,582,912 elements / 8 per thread / 256 per block
    {
        const int threads = (BB * SS * DD) / 8;
        k_convert_x<<<threads / 256, 256, 0, stream>>>(x, Xb);
    }
    // 2) Q/K/V projections: 3 * 1024 * 12 wave-strips, 4 waves per block
    {
        const int waves = 3 * ((BB * SS) / 16) * (DD / 64);    // 36864
        k_qkv_gemm<<<waves / 4, 128, 0, stream>>>(Xb, Wt, Qb, Kb, Vt);
    }
    // 3) Flash attention: one block per 16-query tile
    {
        k_attn<<<BB * (SS / 16), 128, 0, stream>>>(Qb, Kb, Vt, out);
    }
}
